// DotProductAttention_64518998720617
// MI455X (gfx1250) — compile-verified, matrix operand order checked by reading, not run on hardware
//
#include <hip/hip_runtime.h>

// Linear attention with cross-head score normalization, MI455X (gfx1250).
// out = (relu(Q)relu(K)^T / sum_heads(.)) * mask @ V + Q
//
// Phase 1 (d_ws): relu(Q)->bf16, relu(K)->bf16, V^T->bf16 so every WMMA
// operand is two contiguous 16B runs per lane (pure global_load_b128).
// Phase 2: transposed, software-pipelined, M=32 q-rows per wave: K/V
// operand loads are shared by two M-tiles -> 16 WMMAs per 16 operand loads
// (vmem:wmma ~1.7:1), halving L2-side K/V traffic. k-range split in two
// (blockIdx.z) keeps 256 workgroups resident; the cross-head denominator is
// per-(q,k) so the split is exact. Phase 3 combines the halves + residual
// deterministically.

#define BB  2
#define HH  8
#define NNQ 2048
#define NNK 2048
#define DD  64
#define KSPL 2
#define KRNG (NNK / KSPL)

typedef __attribute__((ext_vector_type(16))) __bf16 v16bf;
typedef __attribute__((ext_vector_type(8)))  __bf16 v8bf;
typedef __attribute__((ext_vector_type(8)))  float  v8f;

#define WMMA_BF16(A, B, C) \
  __builtin_amdgcn_wmma_f32_16x16x32_bf16(false, (A), false, (B), (short)0, (C), false, false)

// ---------------- Phase 1 kernels ----------------

__global__ __launch_bounds__(256)
void prep_qk_kernel(const float* __restrict__ q, const float* __restrict__ k,
                    __bf16* __restrict__ qbf, __bf16* __restrict__ kbf) {
  const float* src = blockIdx.y ? k : q;
  __bf16*      dst = blockIdx.y ? kbf : qbf;
  const size_t idx = ((size_t)blockIdx.x * 256 + threadIdx.x) * 8;
  const float4 a = *(const float4*)(src + idx);
  const float4 b = *(const float4*)(src + idx + 4);
  v8bf o;
  o[0] = (__bf16)fmaxf(a.x, 0.f); o[1] = (__bf16)fmaxf(a.y, 0.f);
  o[2] = (__bf16)fmaxf(a.z, 0.f); o[3] = (__bf16)fmaxf(a.w, 0.f);
  o[4] = (__bf16)fmaxf(b.x, 0.f); o[5] = (__bf16)fmaxf(b.y, 0.f);
  o[6] = (__bf16)fmaxf(b.z, 0.f); o[7] = (__bf16)fmaxf(b.w, 0.f);
  *(v8bf*)(dst + idx) = o;
}

__global__ __launch_bounds__(256)
void prep_vt_kernel(const float* __restrict__ v, __bf16* __restrict__ vt) {
  const int bh = blockIdx.y;
  const int d  = threadIdx.x & 63;
  const int k8 = blockIdx.x * 32 + (threadIdx.x >> 6) * 8;
  const float* src = v + (size_t)bh * NNK * DD;
  __bf16*      dst = vt + ((size_t)bh * DD + d) * NNK + k8;
  v8bf o;
#pragma unroll
  for (int j = 0; j < 8; ++j) o[j] = (__bf16)src[(size_t)(k8 + j) * DD + d];
  *(v8bf*)dst = o;
}

// ---------------- Phase 2 kernel ----------------

__device__ __forceinline__ v16bf load_op_bf16(const __bf16* __restrict__ row, int c0) {
  v8bf lo = *(const v8bf*)(row + c0);
  v8bf hi = *(const v8bf*)(row + c0 + 16);
  return __builtin_shufflevector(lo, hi, 0, 1, 2, 3, 4, 5, 6, 7,
                                 8, 9, 10, 11, 12, 13, 14, 15);
}

__device__ __forceinline__ void load_k_ops(v16bf* op, const __bf16* __restrict__ kbbase,
                                           int k0, int row, int koff) {
  const __bf16* kr0 = kbbase + (size_t)(k0 + row) * DD;
  const __bf16* kr1 = kbbase + (size_t)(k0 + 16 + row) * DD;
  op[0] = load_op_bf16(kr0, koff);
  op[1] = load_op_bf16(kr0, 32 + koff);
  op[2] = load_op_bf16(kr1, koff);
  op[3] = load_op_bf16(kr1, 32 + koff);
}

__global__ __launch_bounds__(256)
void linattn_headnorm_kernel(const float* __restrict__ mask,
                             const __bf16* __restrict__ qbf,
                             const __bf16* __restrict__ kbf,
                             const __bf16* __restrict__ vtbf,
                             float* __restrict__ out,
                             float* __restrict__ pout) {
  // 32 score slots per lane (2 q-tiles x 16). Rows padded to 36 floats:
  // 144B rows stay 16B-aligned, b128 bank-group stride 9 (odd).
  __shared__ float sred[HH][32][36];   // 36 KB
  __shared__ float dbuf[32][36];       // denominator totals (32 slots + pad)

  const int lane  = threadIdx.x & 31;
  const int h     = threadIdx.x >> 5;   // wave id == head id
  const int b     = blockIdx.y;
  const int q0    = blockIdx.x * 32;    // two 16-row q-tiles per wave
  const int kbeg  = blockIdx.z * KRNG;

  const int row   = lane & 15;
  const int koff  = (lane >> 4) * 8;

  const size_t bh    = (size_t)b * HH + h;
  const size_t headQ = bh * NNQ;
  const __bf16* kbbase = kbf + bh * NNK * DD;
  const __bf16* vtbase = vtbf + bh * DD * NNK;
  const float*  mbase  = mask + ((size_t)b * NNQ + q0) * NNK;
  float* dst = (blockIdx.z == 0) ? out : pout;

  // Persistent Q B-operands for both q-tiles (pre-relu'd bf16).
  const __bf16* qrowA = qbf + (headQ + q0 + row) * DD;
  const __bf16* qrowB = qbf + (headQ + q0 + 16 + row) * DD;
  const v16bf bqAlo = load_op_bf16(qrowA, koff);
  const v16bf bqAhi = load_op_bf16(qrowA, 32 + koff);
  const v16bf bqBlo = load_op_bf16(qrowB, koff);
  const v16bf bqBhi = load_op_bf16(qrowB, 32 + koff);

  const v8f vzero = {0.f, 0.f, 0.f, 0.f, 0.f, 0.f, 0.f, 0.f};
  v8f acc[2][4];   // out^T accumulators per q-tile
#pragma unroll
  for (int qt = 0; qt < 2; ++qt)
#pragma unroll
    for (int t = 0; t < 4; ++t) acc[qt][t] = vzero;

  auto tile_step = [&](const v16bf* cur, v16bf* nxt, int k0) {
    // Early-issue: shared V operands + both q-tiles' mask rows.
    v16bf av0 = load_op_bf16(vtbase + (size_t)(0 * 16 + row) * NNK, k0 + koff);
    v16bf av1 = load_op_bf16(vtbase + (size_t)(1 * 16 + row) * NNK, k0 + koff);
    v16bf av2 = load_op_bf16(vtbase + (size_t)(2 * 16 + row) * NNK, k0 + koff);
    v16bf av3 = load_op_bf16(vtbase + (size_t)(3 * 16 + row) * NNK, k0 + koff);
    const float* mrA = mbase + (size_t)row * NNK + k0 + koff;
    const float* mrB = mbase + (size_t)(16 + row) * NNK + k0 + koff;
    const float4 mA0 = *(const float4*)(mrA);
    const float4 mA1 = *(const float4*)(mrA + 4);
    const float4 mA2 = *(const float4*)(mrA + 16);
    const float4 mA3 = *(const float4*)(mrA + 20);
    const float4 mB0 = *(const float4*)(mrB);
    const float4 mB1 = *(const float4*)(mrB + 4);
    const float4 mB2 = *(const float4*)(mrB + 16);
    const float4 mB3 = *(const float4*)(mrB + 20);

    // Preload next K tile (wraps inside this z's k-range; branch-free).
    load_k_ops(nxt, kbbase, kbeg + ((k0 - kbeg + 32) & (KRNG - 1)), row, koff);

    // ---- GEMM1^T: S^T[32k x 32q] = relu(K) * relu(Q)^T, K ops shared ----
    v8f s0a = vzero, s1a = vzero, s0b = vzero, s1b = vzero;
    s0a = WMMA_BF16(cur[0], bqAlo, s0a);  s0a = WMMA_BF16(cur[1], bqAhi, s0a);
    s1a = WMMA_BF16(cur[2], bqAlo, s1a);  s1a = WMMA_BF16(cur[3], bqAhi, s1a);
    s0b = WMMA_BF16(cur[0], bqBlo, s0b);  s0b = WMMA_BF16(cur[1], bqBhi, s0b);
    s1b = WMMA_BF16(cur[2], bqBlo, s1b);  s1b = WMMA_BF16(cur[3], bqBhi, s1b);

    // ---- Publish scores: slots 0..7=s0a, 8..15=s1a, 16..23=s0b, 24..31=s1b
#pragma unroll
    for (int i = 0; i < 8; ++i) {
      sred[h][lane][i]      = s0a[i];
      sred[h][lane][8 + i]  = s1a[i];
      sred[h][lane][16 + i] = s0b[i];
      sred[h][lane][24 + i] = s1b[i];
    }
    __syncthreads();  // B1

    // ---- Distributed cross-head reduction: wave h owns slots {4h..4h+3} ----
    {
      const int i0 = 4 * h;
      float p0 = 0.f, p1 = 0.f, p2 = 0.f, p3 = 0.f;
#pragma unroll
      for (int w = 0; w < 8; ++w) {
        const float4 v = *(const float4*)(&sred[w][lane][i0]);
        p0 += v.x; p1 += v.y; p2 += v.z; p3 += v.w;
      }
      float4 pr; pr.x = p0; pr.y = p1; pr.z = p2; pr.w = p3;
      *(float4*)(&dbuf[lane][i0]) = pr;
    }
    __syncthreads();  // B2

    float den[32];
#pragma unroll
    for (int j = 0; j < 8; ++j) {
      const float4 v = *(const float4*)(&dbuf[lane][4 * j]);
      den[4 * j + 0] = v.x; den[4 * j + 1] = v.y;
      den[4 * j + 2] = v.z; den[4 * j + 3] = v.w;
    }

    // ---- P^T = (S^T / den) * mask, in-register, straight into B layout ----
    const float mva[16] = {mA0.x, mA0.y, mA0.z, mA0.w, mA1.x, mA1.y, mA1.z, mA1.w,
                           mA2.x, mA2.y, mA2.z, mA2.w, mA3.x, mA3.y, mA3.z, mA3.w};
    const float mvb[16] = {mB0.x, mB0.y, mB0.z, mB0.w, mB1.x, mB1.y, mB1.z, mB1.w,
                           mB2.x, mB2.y, mB2.z, mB2.w, mB3.x, mB3.y, mB3.z, mB3.w};
    v16bf bpa, bpb;
#pragma unroll
    for (int r = 0; r < 8; ++r) {
      bpa[r]     = (__bf16)(s0a[r] * __builtin_amdgcn_rcpf(den[r])      * mva[r]);
      bpa[8 + r] = (__bf16)(s1a[r] * __builtin_amdgcn_rcpf(den[8 + r])  * mva[8 + r]);
      bpb[r]     = (__bf16)(s0b[r] * __builtin_amdgcn_rcpf(den[16 + r]) * mvb[r]);
      bpb[8 + r] = (__bf16)(s1b[r] * __builtin_amdgcn_rcpf(den[24 + r]) * mvb[8 + r]);
    }

    // ---- GEMM2^T: out^T += V^T * P^T, V ops shared by both q-tiles ----
    acc[0][0] = WMMA_BF16(av0, bpa, acc[0][0]);
    acc[0][1] = WMMA_BF16(av1, bpa, acc[0][1]);
    acc[0][2] = WMMA_BF16(av2, bpa, acc[0][2]);
    acc[0][3] = WMMA_BF16(av3, bpa, acc[0][3]);
    acc[1][0] = WMMA_BF16(av0, bpb, acc[1][0]);
    acc[1][1] = WMMA_BF16(av1, bpb, acc[1][1]);
    acc[1][2] = WMMA_BF16(av2, bpb, acc[1][2]);
    acc[1][3] = WMMA_BF16(av3, bpb, acc[1][3]);

    // Light prefetch two tiles ahead (global_prefetch_b8).
    const int kp = kbeg + ((k0 - kbeg + 64) & (KRNG - 1));
    __builtin_prefetch(kbbase + (size_t)(kp + lane) * DD, 0, 1);
    __builtin_prefetch(vtbase + (size_t)lane * NNK + kp, 0, 1);
    __builtin_prefetch(vtbase + (size_t)(32 + lane) * NNK + kp, 0, 1);
  };

  v16bf bufA[4], bufB[4];
  load_k_ops(bufA, kbbase, kbeg, row, koff);
  for (int k0 = kbeg; k0 < kbeg + KRNG; k0 += 64) {
    tile_step(bufA, bufB, k0);
    tile_step(bufB, bufA, k0 + 32);
  }

  // ---- Epilogue: store partial out^T (residual added in combine pass) ----
#pragma unroll
  for (int qt = 0; qt < 2; ++qt) {
    float* orow = dst + (headQ + q0 + qt * 16 + row) * DD;
#pragma unroll
    for (int t = 0; t < 4; ++t)
#pragma unroll
      for (int r = 0; r < 8; ++r)
        orow[t * 16 + koff + r] = acc[qt][t][r];
  }
}

// ---------------- Phase 3: combine k-halves + residual ----------------

__global__ __launch_bounds__(256)
void combine_kernel(float* __restrict__ out, const float* __restrict__ part,
                    const float* __restrict__ query) {
  const size_t i = ((size_t)blockIdx.x * 256 + threadIdx.x) * 4;
  float4 o = *(const float4*)(out + i);
  const float4 p = *(const float4*)(part + i);
  const float4 q = *(const float4*)(query + i);
  o.x += p.x + q.x; o.y += p.y + q.y; o.z += p.z + q.z; o.w += p.w + q.w;
  *(float4*)(out + i) = o;
}

extern "C" void kernel_launch(void* const* d_in, const int* in_sizes, int n_in,
                              void* d_out, int out_size, void* d_ws, size_t ws_size,
                              hipStream_t stream) {
  (void)in_sizes; (void)n_in; (void)out_size; (void)ws_size;
  const float* query = (const float*)d_in[0];
  const float* key   = (const float*)d_in[1];
  const float* value = (const float*)d_in[2];
  const float* mask  = (const float*)d_in[3];
  float* out = (float*)d_out;

  const size_t nElem = (size_t)BB * HH * NNQ * DD;    // 2M per tensor
  __bf16* qbf  = (__bf16*)d_ws;                       // 4 MiB
  __bf16* kbf  = qbf + nElem;                         // 4 MiB
  __bf16* vtbf = kbf + nElem;                         // 4 MiB
  float*  pout = (float*)(vtbf + nElem);              // 8 MiB (20 MiB total)

  {
    dim3 grid((unsigned)(nElem / (256 * 8)), 2);
    prep_qk_kernel<<<grid, 256, 0, stream>>>(query, key, qbf, kbf);
  }
  {
    dim3 grid(NNK / 32, BB * HH);
    prep_vt_kernel<<<grid, 256, 0, stream>>>(value, vtbf);
  }
  {
    dim3 grid(NNQ / 32, BB, KSPL);                    // (64, 2, 2) = 256 WGs
    linattn_headnorm_kernel<<<grid, 256, 0, stream>>>(mask, qbf, kbf, vtbf, out, pout);
  }
  {
    dim3 grid((unsigned)(nElem / (256 * 4)));
    combine_kernel<<<grid, 256, 0, stream>>>(out, pout, query);
  }
}
